// MultiHeadAttention_63677185130528
// MI455X (gfx1250) — matrix operand order repaired and checked by reading, not compiled, not run on hardware
//
#include <hip/hip_runtime.h>

typedef __bf16 bf16;
typedef __attribute__((ext_vector_type(16))) __bf16 v16bf;
typedef __attribute__((ext_vector_type(8)))  float  v8f;
typedef __attribute__((ext_vector_type(4)))  int    v4i;

#define BM 128
#define BN 128
#define BKK 32
#define BKP 40   // padded K stride in LDS (elements); row byte stride 80 (16B multiple)

// Low 32 bits of a generic pointer into the LDS aperture == LDS byte offset
__device__ __forceinline__ unsigned lds_off(const void* p) {
    return (unsigned)(size_t)p;
}

// CDNA5 async global->LDS copy (ASYNCcnt tracked, no VGPR roundtrip)
__device__ __forceinline__ void async_copy_b128(void* lds_dst, const void* gsrc) {
    asm volatile("global_load_async_to_lds_b128 %0, %1, off"
                 :: "v"(lds_off(lds_dst)),
                    "v"((unsigned long long)(size_t)gsrc)
                 : "memory");
}

__device__ __forceinline__ void wait_async0() {
    asm volatile("s_wait_asynccnt 0x0" ::: "memory");
}

__device__ __forceinline__ v16bf make_frag(const bf16* lo, const bf16* hi) {
    union U { v16bf v; v4i q[2]; } u;
    u.q[0] = *(const v4i*)lo;
    u.q[1] = *(const v4i*)hi;
    return u.v;
}

// C[M,N] = alpha * A[M,K] @ op(B) + bias ;  op(B)=B[K,N] (TRANSB=0) or B[N,K]^T (TRANSB=1)
// A,B bf16 row-major; C is OutT (float or bf16). grid.z = batch with given batch strides.
template<typename OutT, bool TRANSB>
__global__ __launch_bounds__(256) void gemm_bf16_wmma(
    const bf16* __restrict__ A, const bf16* __restrict__ Bm, OutT* __restrict__ C,
    const float* __restrict__ bias, float alpha,
    int M, int N, int K, int lda, int ldb, int ldc,
    long long sAb, long long sBb, long long sCb)
{
    __shared__ bf16 Alds[2][BM][BKP] __attribute__((aligned(16)));
    __shared__ bf16 Blds[2][BN][BKP] __attribute__((aligned(16)));   // Blds[n][k]

    const int tid  = threadIdx.x;
    const int lane = tid & 31;
    const int wave = tid >> 5;
    const int wm   = wave >> 1;          // 0..3 -> 32 rows each
    const int wn   = wave & 1;           // 0..1 -> 64 cols each
    const int hf   = lane >> 4;          // lane half
    const int l16  = lane & 15;

    const int blockRow = blockIdx.y * BM;
    const int blockCol = blockIdx.x * BN;
    const long long z = blockIdx.z;

    A  += z * sAb;
    Bm += z * sBb;
    C  += z * sCb;

    v8f acc[2][4];
    #pragma unroll
    for (int mt = 0; mt < 2; ++mt)
        #pragma unroll
        for (int nt = 0; nt < 4; ++nt)
            acc[mt][nt] = v8f{};

    const int kOffA = hf * 8;   // A frag: K {0..7,16..23} or {8..15,24..31}
    const int kOffB = hf * 8;   // B frag: K {0..7,16..23} or {8..15,24..31}

    // ---- staging helpers -------------------------------------------------
    // A tile (128x32): 2 async b128 per thread
    auto stageA = [&](int buf, int k0) {
        int r  = tid >> 1;
        int c0 = (tid & 1) * 16;
        const bf16* src = A + (long long)(blockRow + r) * lda + k0 + c0;
        async_copy_b128(&Alds[buf][r][c0],     src);
        async_copy_b128(&Alds[buf][r][c0 + 8], src + 8);
    };
    // B tile, TRANSB: B is [N,K] row-major -> straight async copy into Blds[n][k]
    auto stageB_T = [&](int buf, int k0) {
        int n  = tid >> 1;
        int c0 = (tid & 1) * 16;
        const bf16* src = Bm + (long long)(blockCol + n) * ldb + k0 + c0;
        async_copy_b128(&Blds[buf][n][c0],     src);
        async_copy_b128(&Blds[buf][n][c0 + 8], src + 8);
    };
    // B tile, normal: B is [K,N]; phase 1 issues coalesced global loads of two
    // adjacent K rows, phase 2 scatters packed {k,k+1} pairs as b32 into LDS.
    v4i breg0{}, breg1{};
    auto loadB_N = [&](int k0) {
        int kk = (tid >> 4) * 2;        // 0,2,..,30
        int n0 = (tid & 15) * 8;        // 0,8,..,120 (coalesced along N)
        const bf16* s0 = Bm + (long long)(k0 + kk) * ldb + blockCol + n0;
        breg0 = *(const v4i*)s0;
        breg1 = *(const v4i*)(s0 + ldb);
    };
    auto storeB_N = [&](int buf) {
        int kk = (tid >> 4) * 2;
        int n0 = (tid & 15) * 8;
        unsigned short lo[8], hi[8];
        *(v4i*)lo = breg0;
        *(v4i*)hi = breg1;
        #pragma unroll
        for (int i = 0; i < 8; ++i) {
            unsigned pk = (unsigned)lo[i] | ((unsigned)hi[i] << 16);
            *(unsigned*)&Blds[buf][n0 + i][kk] = pk;   // kk even -> 4B aligned
        }
    };

    // ---- prologue: fill buffer 0 ----------------------------------------
    const int nk = K / BKK;
    stageA(0, 0);
    if (TRANSB) {
        stageB_T(0, 0);
    } else {
        loadB_N(0);
        storeB_N(0);
    }
    wait_async0();
    __syncthreads();

    // ---- double-buffered main loop: one barrier per K step ---------------
    for (int t = 0; t < nk; ++t) {
        const int cur = t & 1;
        const int nxt = cur ^ 1;
        const bool more = (t + 1 < nk);

        if (more) {
            stageA(nxt, (t + 1) * BKK);
            if (TRANSB) stageB_T(nxt, (t + 1) * BKK);
            else        loadB_N((t + 1) * BKK);
        }

        // ---- fragments per ISA VGPR layouts ----
        v16bf afrag[2];
        #pragma unroll
        for (int mt = 0; mt < 2; ++mt) {
            int r = wm * 32 + mt * 16 + l16;
            afrag[mt] = make_frag(&Alds[cur][r][kOffA], &Alds[cur][r][kOffA + 16]);
        }
        v16bf bfrag[4];
        #pragma unroll
        for (int nt = 0; nt < 4; ++nt) {
            int n = wn * 64 + nt * 16 + l16;
            bfrag[nt] = make_frag(&Blds[cur][n][kOffB], &Blds[cur][n][kOffB + 16]);
        }

        #pragma unroll
        for (int mt = 0; mt < 2; ++mt)
            #pragma unroll
            for (int nt = 0; nt < 4; ++nt)
                acc[mt][nt] = __builtin_amdgcn_wmma_f32_16x16x32_bf16(
                    false, afrag[mt], false, bfrag[nt],
                    (short)0, acc[mt][nt], false, false);

        if (more) {
            if (!TRANSB) storeB_N(nxt);
            wait_async0();
        }
        __syncthreads();
    }

    // ---- epilogue: C/D layout -> lane<16: M=r,N=lane ; lane>=16: M=8+r ----
    #pragma unroll
    for (int mt = 0; mt < 2; ++mt) {
        #pragma unroll
        for (int nt = 0; nt < 4; ++nt) {
            int n     = blockCol + wn * 64 + nt * 16 + l16;
            int mBase = blockRow + wm * 32 + mt * 16 + hf * 8;
            float bv = bias ? bias[n] : 0.0f;
            #pragma unroll
            for (int r = 0; r < 8; ++r) {
                float v = acc[mt][nt][r] * alpha + bv;
                C[(long long)(mBase + r) * ldc + n] = (OutT)v;
            }
        }
    }
}

__global__ void f32_to_bf16_kernel(const float* __restrict__ in, bf16* __restrict__ out,
                                   long long n) {
    long long i = (long long)blockIdx.x * blockDim.x + threadIdx.x;
    long long stride = (long long)gridDim.x * blockDim.x;
    for (; i < n; i += stride) out[i] = (bf16)in[i];
}

// One block per row; T == 2048 == 256 threads * 8 elems.
__global__ __launch_bounds__(256) void softmax_rows_kernel(
    const float* __restrict__ S, bf16* __restrict__ W, int T)
{
    const int row = blockIdx.x;
    const int tid = threadIdx.x;
    const float* srow = S + (long long)row * T;

    float v[8];
    #pragma unroll
    for (int i = 0; i < 8; ++i) v[i] = srow[tid * 8 + i];

    __shared__ float red[256];
    float m = v[0];
    #pragma unroll
    for (int i = 1; i < 8; ++i) m = fmaxf(m, v[i]);
    red[tid] = m; __syncthreads();
    for (int s = 128; s > 0; s >>= 1) {
        if (tid < s) red[tid] = fmaxf(red[tid], red[tid + s]);
        __syncthreads();
    }
    const float rmax = red[0]; __syncthreads();

    float sum = 0.0f;
    #pragma unroll
    for (int i = 0; i < 8; ++i) { v[i] = __expf(v[i] - rmax); sum += v[i]; }
    red[tid] = sum; __syncthreads();
    for (int s = 128; s > 0; s >>= 1) {
        if (tid < s) red[tid] += red[tid + s];
        __syncthreads();
    }
    const float inv = 1.0f / red[0];

    bf16* wrow = W + (long long)row * T;
    #pragma unroll
    for (int i = 0; i < 8; ++i) wrow[tid * 8 + i] = (bf16)(v[i] * inv);
}

extern "C" void kernel_launch(void* const* d_in, const int* in_sizes, int n_in,
                              void* d_out, int out_size, void* d_ws, size_t ws_size,
                              hipStream_t stream) {
    const float* x      = (const float*)d_in[0];
    const float* W_attn = (const float*)d_in[1];
    const float* b_attn = (const float*)d_in[2];
    const float* W_proj = (const float*)d_in[3];
    const float* b_proj = (const float*)d_in[4];
    float* out = (float*)d_out;

    const int Bb = 4, S = 2048, E = 1024, T = 2048;
    const int M = Bb * S;                       // 8192
    const long long nx  = (long long)M * E;
    const long long nwa = (long long)E * 3 * E;
    const long long nwp = (long long)E * E;

    char* p = (char*)d_ws;
    auto carve = [&p](size_t bytes) {
        void* r = (void*)p;
        p += (bytes + 255) & ~(size_t)255;
        return r;
    };
    bf16*  xb  = (bf16*)carve(nx * 2);                       // x in bf16
    bf16*  wab = (bf16*)carve(nwa * 2);                      // W_attn bf16
    bf16*  wpb = (bf16*)carve(nwp * 2);                      // W_proj bf16
    bf16*  cb  = (bf16*)carve((long long)M * 3 * E * 2);     // qkv bf16 [8192,3072]
    float* sc  = (float*)carve((long long)Bb * S * T * 4);   // scores f32
    bf16*  wb  = (bf16*)carve((long long)Bb * S * T * 2);    // softmax weights bf16
    bf16*  ab  = (bf16*)carve((long long)M * E * 2);         // attn bf16

    dim3 blk(256);
    f32_to_bf16_kernel<<<2048, blk, 0, stream>>>(x, xb, nx);
    f32_to_bf16_kernel<<<2048, blk, 0, stream>>>(W_attn, wab, nwa);
    f32_to_bf16_kernel<<<1024, blk, 0, stream>>>(W_proj, wpb, nwp);

    // 1) qkv = x @ W_attn + b_attn  -> cb (bf16)
    gemm_bf16_wmma<bf16, false><<<dim3(3 * E / BN, M / BM, 1), blk, 0, stream>>>(
        xb, wab, cb, b_attn, 1.0f,
        M, 3 * E, E, E, 3 * E, 3 * E, 0, 0, 0);

    // 2) scores = (q @ k^T) * 1/sqrt(64)  -> sc (f32), per batch
    gemm_bf16_wmma<float, true><<<dim3(T / BN, S / BM, Bb), blk, 0, stream>>>(
        cb /*q*/, cb + E /*k*/, sc, nullptr, 0.125f,
        S, T, E, 3 * E, 3 * E, T,
        (long long)S * 3 * E, (long long)S * 3 * E, (long long)S * T);

    // 3) softmax rows -> wb (bf16)
    softmax_rows_kernel<<<Bb * S, blk, 0, stream>>>(sc, wb, T);

    // 4) attn = weights @ v -> ab (bf16), per batch
    gemm_bf16_wmma<bf16, false><<<dim3(E / BN, S / BM, Bb), blk, 0, stream>>>(
        wb, cb + 2 * E /*v*/, ab, nullptr, 1.0f,
        S, E, T, T, 3 * E, E,
        (long long)S * T, (long long)S * 3 * E, (long long)S * E);

    // 5) out = attn @ W_proj + b_proj -> d_out (f32)
    gemm_bf16_wmma<float, false><<<dim3(E / BN, M / BM, 1), blk, 0, stream>>>(
        ab, wpb, out, b_proj, 1.0f,
        M, E, E, E, E, E, 0, 0, 0);
}
